// ConvmambaProj_16561393894006
// MI455X (gfx1250) — matrix operand order repaired and checked by reading, not compiled, not run on hardware
//
#include <hip/hip_runtime.h>
#include <hip/hip_bf16.h>
#include <math.h>

typedef __bf16 bf16_t;
typedef __attribute__((ext_vector_type(16))) __bf16 v16bf;
typedef __attribute__((ext_vector_type(8)))  float  v8f;
typedef __attribute__((ext_vector_type(4)))  unsigned int v4u;
typedef __attribute__((ext_vector_type(8)))  int v8i;
typedef __attribute__((ext_vector_type(4)))  int v4i;

#define D_MODEL 512
#define D_INNER 1024
#define D_STATE 16
#define DT_RANK 32
#define BATCH   4
#define SEQLEN  2304
#define NDIR    8
#define XZ_LD   (5*D_INNER)      // 5120
#define OUTF_LD (3*D_INNER)      // 3072
#define CAT_LD  (4*D_MODEL)      // 2048

__device__ __forceinline__ void splitf(float v, bf16_t& h, bf16_t& l) {
    h = (bf16_t)v;
    l = (bf16_t)(v - (float)h);
}

// ---------------------------------------------------------------------------
// TDM: issue a 2-D tensor_load_to_lds of (rows x 64B) with global row stride
// `stride8` (in 8-byte units). D# built per ISA 8.3/8.4; groups 2/3 zero (2D).
// Per-wave instruction (EXEC ignored); caller guards so only wave 0 issues.
// ---------------------------------------------------------------------------
__device__ __forceinline__ void tdm_load_tile(unsigned int lds_addr,
                                              const bf16_t* gsrc,
                                              unsigned int rows,
                                              unsigned int stride8) {
    unsigned long long ga = (unsigned long long)(uintptr_t)gsrc;
    v4u g0 = { 1u,                                  // count=1, user mode
               lds_addr,                            // LDS byte address
               (unsigned int)ga,                    // global_addr[31:0]
               (unsigned int)((ga >> 32) & 0x01FFFFFFu) | (2u << 30) }; // [56:32] | type=2
    const unsigned int td0 = 8u, til0 = 8u;         // 8 x 8B = 64B per row
    v8i g1 = {0,0,0,0,0,0,0,0};
    g1[0] = (int)(3u << 16);                        // data_size = 8B
    g1[1] = (int)((td0 & 0xFFFFu) << 16);           // tensor_dim0[15:0]
    g1[2] = (int)((td0 >> 16) | ((rows & 0xFFFFu) << 16));   // td0 hi | tensor_dim1 lo
    g1[3] = (int)((rows >> 16) | (til0 << 16));     // tensor_dim1 hi | tile_dim0
    g1[4] = (int)(rows & 0xFFFFu);                  // tile_dim1 (tile_dim2 = 0)
    g1[5] = (int)stride8;                           // tensor_dim0_stride[31:0]
    v4i gz = {0, 0, 0, 0};
#if __clang_major__ >= 23
    v8i gz8 = {0,0,0,0,0,0,0,0};
    __builtin_amdgcn_tensor_load_to_lds(g0, g1, gz, gz, gz8, 0);
#else
    __builtin_amdgcn_tensor_load_to_lds(g0, g1, gz, gz, 0);
#endif
}

__device__ __forceinline__ unsigned int lds_off32(const void* p) {
    return (unsigned int)(uintptr_t)p;              // LDS addr = generic addr[31:0]
}

// ---------------------------------------------------------------------------
// Split an f32 array into bf16 hi/lo pair (bf16x3 GEMM inputs)
// ---------------------------------------------------------------------------
__global__ __launch_bounds__(256) void split_kernel(const float* __restrict__ in,
                                                    bf16_t* __restrict__ hi,
                                                    bf16_t* __restrict__ lo, int n) {
    int i = blockIdx.x * 256 + threadIdx.x;
    if (i < n) { bf16_t h, l; splitf(in[i], h, l); hi[i] = h; lo[i] = l; }
}

// ---------------------------------------------------------------------------
// Generic batched bf16x3 WMMA GEMM with TDM double-buffered LDS staging.
//   C[b][n, m] = sum_k A[m, k] * B[b][n, k]   (+ optional bias[m])
// Tile (16*NWAVES) x 64, 32*NWAVES threads; wave w owns M rows [16w, 16w+16).
// Wave 0 DMAs K-tile k+1 (4 descriptors: A-hi/lo, B-hi/lo) into the spare
// buffer while all waves run 12 WMMAs per wave on K-tile k; s_wait_tensorcnt
// + workgroup barrier publish the LDS.
// Fragments load as contiguous ds b128 chunks:
//   A lane(g,m):  K in [8g..8g+7] U [16+8g..23+8g]
//   B lane(g,n):  K in [8g..8g+7] U [16+8g..23+8g]
// Epilogue transposes through padded LDS -> (n, m) layout, optional f32 and
// split-bf16 outputs (inputs of the next GEMM in the chain).
// ---------------------------------------------------------------------------
union FragU { uint4 q[2]; v16bf v; };

template <int NWAVES>
struct GemmSmem {
    union {
        struct {
            bf16_t ah[NWAVES * 16 * 32];
            bf16_t al[NWAVES * 16 * 32];
            bf16_t bh[64 * 32];
            bf16_t bl[64 * 32];
        } buf[2];
        float c[64 * (NWAVES * 16 + 4)];
    };
};

template <int NWAVES>
__global__ __launch_bounds__(32 * NWAVES) void gemm_bf16x3_kernel(
    const bf16_t* __restrict__ Ahi, const bf16_t* __restrict__ Alo, int lda,
    const bf16_t* __restrict__ Bhi, const bf16_t* __restrict__ Blo, int ldb, size_t strideB,
    float*  Cf,  int ldcf, size_t strideCf,
    bf16_t* Chi, bf16_t* Clo, int ldcb, size_t strideCb,
    const float* __restrict__ biasM,
    int K)
{
    constexpr int BM  = 16 * NWAVES;
    constexpr int NTH = 32 * NWAVES;
    constexpr int CP  = BM + 4;            // padded C stride (16B-aligned rows)
    __shared__ GemmSmem<NWAVES> u;

    const int tid  = threadIdx.x;
    const int wave = tid >> 5;
    const int lane = tid & 31;
    const int g    = lane >> 4;
    const int mr   = lane & 15;

    const int n0 = blockIdx.x * 64;
    const int m0 = blockIdx.y * BM;
    const int b  = blockIdx.z;

    const bf16_t* bhi = Bhi + (size_t)b * strideB;
    const bf16_t* blo = Blo + (size_t)b * strideB;

    v8f acc[4] = {v8f{0}, v8f{0}, v8f{0}, v8f{0}};

    const int nk = K / 32;

    // issue K-tile `kt` into buffer `bi` (wave 0 only; 4 TDM descriptors)
    auto issue = [&](int kt, int bi) {
        const int k0 = kt * 32;
        tdm_load_tile(lds_off32(u.buf[bi].ah), &Ahi[(size_t)m0 * lda + k0],
                      (unsigned)BM, (unsigned)(lda >> 2));
        tdm_load_tile(lds_off32(u.buf[bi].al), &Alo[(size_t)m0 * lda + k0],
                      (unsigned)BM, (unsigned)(lda >> 2));
        tdm_load_tile(lds_off32(u.buf[bi].bh), &bhi[(size_t)n0 * ldb + k0],
                      64u, (unsigned)(ldb >> 2));
        tdm_load_tile(lds_off32(u.buf[bi].bl), &blo[(size_t)n0 * ldb + k0],
                      64u, (unsigned)(ldb >> 2));
    };

    if (wave == 0) issue(0, 0);

    for (int ki = 0; ki < nk; ++ki) {
        const int cur = ki & 1;
        if (wave == 0) {
            if (ki + 1 < nk) {
                issue(ki + 1, cur ^ 1);
                __builtin_amdgcn_s_wait_tensorcnt(4);   // tile ki complete (in-order)
            } else {
                __builtin_amdgcn_s_wait_tensorcnt(0);
            }
        }
        __syncthreads();    // publish tile ki

        const int arow = wave * 16 + mr;
        FragU afh, afl;
        afh.q[0] = *(uint4*)&u.buf[cur].ah[arow * 32 + 8 * g];
        afh.q[1] = *(uint4*)&u.buf[cur].ah[arow * 32 + 16 + 8 * g];
        afl.q[0] = *(uint4*)&u.buf[cur].al[arow * 32 + 8 * g];
        afl.q[1] = *(uint4*)&u.buf[cur].al[arow * 32 + 16 + 8 * g];

        #pragma unroll
        for (int j = 0; j < 4; ++j) {
            const int nc = j * 16 + mr;
            FragU bfh, bfl;
            bfh.q[0] = *(uint4*)&u.buf[cur].bh[nc * 32 + 8 * g];
            bfh.q[1] = *(uint4*)&u.buf[cur].bh[nc * 32 + 16 + 8 * g];
            bfl.q[0] = *(uint4*)&u.buf[cur].bl[nc * 32 + 8 * g];
            bfl.q[1] = *(uint4*)&u.buf[cur].bl[nc * 32 + 16 + 8 * g];
            acc[j] = __builtin_amdgcn_wmma_f32_16x16x32_bf16(false, afh.v, false, bfh.v,
                                                             (short)0, acc[j], false, false);
            acc[j] = __builtin_amdgcn_wmma_f32_16x16x32_bf16(false, afh.v, false, bfl.v,
                                                             (short)0, acc[j], false, false);
            acc[j] = __builtin_amdgcn_wmma_f32_16x16x32_bf16(false, afl.v, false, bfh.v,
                                                             (short)0, acc[j], false, false);
        }
        __syncthreads();    // all reads of buf[cur] done before it is re-DMAed
    }

    // Epilogue: transpose through LDS so output is (n, m) with coalesced stores.
    #pragma unroll
    for (int j = 0; j < 4; ++j) {
        const int n_l = j * 16 + mr;
        #pragma unroll
        for (int r = 0; r < 8; ++r) {
            const int m_l = wave * 16 + g * 8 + r;   // D layout: VGPR r -> M = r + 8g
            u.c[n_l * CP + m_l] = acc[j][r];
        }
    }
    __syncthreads();

    for (int chunk = tid; chunk < 64 * (BM / 4); chunk += NTH) {
        int n_l = chunk / (BM / 4);
        int m4  = (chunk % (BM / 4)) * 4;
        float4 v = *(float4*)&u.c[n_l * CP + m4];
        if (biasM) {
            v.x += biasM[m0 + m4 + 0]; v.y += biasM[m0 + m4 + 1];
            v.z += biasM[m0 + m4 + 2]; v.w += biasM[m0 + m4 + 3];
        }
        size_t n = (size_t)(n0 + n_l);
        if (Cf)  *(float4*)&Cf[(size_t)b * strideCf + n * ldcf + m0 + m4] = v;
        if (Chi) {
            union { uint2 u2; bf16_t bb[4]; } ph, pl;
            splitf(v.x, ph.bb[0], pl.bb[0]); splitf(v.y, ph.bb[1], pl.bb[1]);
            splitf(v.z, ph.bb[2], pl.bb[2]); splitf(v.w, ph.bb[3], pl.bb[3]);
            size_t o = (size_t)b * strideCb + n * ldcb + m0 + m4;
            *(uint2*)&Chi[o] = ph.u2;
            *(uint2*)&Clo[o] = pl.u2;
        }
    }
}

// ---------------------------------------------------------------------------
// Causal depthwise conv (k=4) + bias + SiLU, building forward & flipped-
// backward streams from xz. Writes x_act f32 (bd, l, d) + hi/lo split.
// grid = (SEQLEN, NDIR), block 256.
// ---------------------------------------------------------------------------
__global__ __launch_bounds__(256) void conv_prep_kernel(
    const float* __restrict__ xz,          // (B, L, 5120)
    const float* __restrict__ cw,          // (1024, 1, 4)
    const float* __restrict__ cb,          // (1024)
    float* __restrict__ xact,
    bf16_t* __restrict__ xact_hi, bf16_t* __restrict__ xact_lo)
{
    const int l  = blockIdx.x;
    const int bd = blockIdx.y;
    const bool fwd = bd < BATCH;
    const int b    = fwd ? bd : bd - BATCH;
    const int xoff = fwd ? 0 : 2 * D_INNER;

    for (int d = threadIdx.x; d < D_INNER; d += 256) {
        float w0 = cw[d*4+0], w1 = cw[d*4+1], w2 = cw[d*4+2], w3 = cw[d*4+3];
        float acc = cb[d];
        #pragma unroll
        for (int t = 0; t < 4; ++t) {
            int lt = l - 3 + t;
            if (lt >= 0) {
                int pos = fwd ? lt : (SEQLEN - 1 - lt);
                float xv = xz[((size_t)b * SEQLEN + pos) * XZ_LD + xoff + d];
                float wt = (t == 0) ? w0 : (t == 1) ? w1 : (t == 2) ? w2 : w3;
                acc += xv * wt;
            }
        }
        float s = acc / (1.f + __expf(-acc));     // SiLU
        size_t o = ((size_t)bd * SEQLEN + l) * D_INNER + d;
        xact[o] = s;
        bf16_t h, lo; splitf(s, h, lo);
        xact_hi[o] = h; xact_lo[o] = lo;
    }
}

// ---------------------------------------------------------------------------
// Selective scan: one thread per (direction, channel). Software-pipelined:
// step l+1's operands are loaded before step l's recurrence is evaluated.
// grid = (D_INNER/256, NDIR), block 256.
// ---------------------------------------------------------------------------
__global__ __launch_bounds__(256) void scan_kernel(
    const float* __restrict__ delta_raw,   // (NDIR, L, 1024)
    const float* __restrict__ dt_bias,     // (1024)
    const float* __restrict__ xact,        // (NDIR, L, 1024)
    const float* __restrict__ xdbl,        // (NDIR, L, 64): [dt(32) | B(16) | C(16)]
    const float* __restrict__ A_log,       // (1024, 16)
    const float* __restrict__ Dp,          // (1024)
    const float* __restrict__ xz,          // (B, L, 5120)
    bf16_t* __restrict__ outf_hi, bf16_t* __restrict__ outf_lo)   // (B, L, 3072)
{
    const int d  = blockIdx.x * 256 + threadIdx.x;
    const int bd = blockIdx.y;
    const bool fwd = bd < BATCH;
    const int b    = fwd ? bd : bd - BATCH;
    const int zcol = fwd ? (D_INNER + d) : (3 * D_INNER + d);
    const int ocol = fwd ? d : (D_INNER + d);

    float Ac[D_STATE];
    #pragma unroll
    for (int j = 0; j < D_STATE; ++j) Ac[j] = -__expf(A_log[d * D_STATE + j]);
    float h[D_STATE];
    #pragma unroll
    for (int j = 0; j < D_STATE; ++j) h[j] = 0.f;

    const float Dd = Dp[d];
    const float db = dt_bias[d];

    size_t row = (size_t)bd * SEQLEN;
    float draw_c = delta_raw[row * D_INNER + d];
    float u_c    = xact[row * D_INNER + d];
    float z_c    = xz[((size_t)b * SEQLEN + (fwd ? 0 : SEQLEN - 1)) * XZ_LD + zcol];

    for (int l = 0; l < SEQLEN; ++l) {
        float draw_n = 0.f, u_n = 0.f, z_n = 0.f;
        if (l + 1 < SEQLEN) {
            size_t rn = (size_t)bd * SEQLEN + (l + 1);
            draw_n = delta_raw[rn * D_INNER + d];
            u_n    = xact[rn * D_INNER + d];
            int pn = fwd ? (l + 1) : (SEQLEN - 2 - l);
            z_n    = xz[((size_t)b * SEQLEN + pn) * XZ_LD + zcol];
        }
        size_t rc = (size_t)bd * SEQLEN + l;
        const float* bc = &xdbl[rc * 64 + DT_RANK];

        float draw = draw_c + db;
        float delta = (draw > 20.f) ? draw : log1pf(__expf(draw));
        float du = delta * u_c;
        float y = 0.f;
        #pragma unroll
        for (int j = 0; j < D_STATE; ++j) {
            float dA = __expf(delta * Ac[j]);
            h[j] = dA * h[j] + du * bc[j];
            y += h[j] * bc[D_STATE + j];
        }
        y += Dd * u_c;
        float zv = z_c;
        y *= zv / (1.f + __expf(-zv));
        int pos = fwd ? l : (SEQLEN - 1 - l);
        size_t o = ((size_t)b * SEQLEN + pos) * OUTF_LD + ocol;
        bf16_t hi, lo; splitf(y, hi, lo);
        outf_hi[o] = hi; outf_lo[o] = lo;

        draw_c = draw_n; u_c = u_n; z_c = z_n;
    }
}

// ---------------------------------------------------------------------------
// Mean of xc over L for the global gate.  grid = BATCH, block = 512 (c=tid).
// ---------------------------------------------------------------------------
__global__ __launch_bounds__(512) void xc_mean_kernel(const float* __restrict__ xz,
                                                      float* __restrict__ mean) {
    const int b = blockIdx.x;
    const int c = threadIdx.x;
    float s = 0.f;
    for (int l = 0; l < SEQLEN; ++l)
        s += xz[((size_t)b * SEQLEN + l) * XZ_LD + 4 * D_INNER + D_MODEL + c];
    mean[b * D_MODEL + c] = s * (1.f / SEQLEN);
}

// ---------------------------------------------------------------------------
// Conv-branch: dilated convs (pre-affine), local conv + GELU, global gate,
// channel LayerNorm at each (b,l); outputs normalized cat as bf16 hi/lo.
// grid = (SEQLEN, BATCH), block 256; each thread handles 8 of 2048 channels.
// ---------------------------------------------------------------------------
__global__ __launch_bounds__(256) void cb_cat_ln_kernel(
    const float* __restrict__ xz,
    const float* __restrict__ lw, const float* __restrict__ lb,     // local (512,1,3),(512)
    const float* __restrict__ gw, const float* __restrict__ gb,     // global (512),(512)
    const float* __restrict__ pw, const float* __restrict__ pb,     // pre (3,512),(3,512)
    const float* __restrict__ dw, const float* __restrict__ dbias,  // dil (3,512,1,3),(3,512)
    const float* __restrict__ ln_g, const float* __restrict__ ln_b, // (2048)
    const float* __restrict__ xc_mean,                              // (B,512)
    bf16_t* __restrict__ cat_hi, bf16_t* __restrict__ cat_lo)       // (B,L,2048)
{
    __shared__ float vals[CAT_LD];
    __shared__ float rs[256], rq[256];
    const int l = blockIdx.x;
    const int b = blockIdx.y;
    const int tid = threadIdx.x;
    const int DIL[3] = {1, 2, 4};

    float s = 0.f, q = 0.f;
    #pragma unroll
    for (int i = 0; i < 8; ++i) {
        int cc   = tid + 256 * i;
        int comp = cc >> 9;
        int c    = cc & (D_MODEL - 1);
        float val;
        if (comp < 3) {
            int dil = DIL[comp];
            float p_w = pw[comp * D_MODEL + c];
            float p_b = pb[comp * D_MODEL + c];
            float acc = dbias[comp * D_MODEL + c];
            #pragma unroll
            for (int j = 0; j < 3; ++j) {
                int lt = l + (j - 1) * dil;
                if (lt >= 0 && lt < SEQLEN) {
                    float xa = xz[((size_t)b * SEQLEN + lt) * XZ_LD + 4 * D_INNER + c];
                    acc += (p_w * xa + p_b) * dw[(comp * D_MODEL + c) * 3 + j];
                }
            }
            val = acc;
        } else {
            float acc = lb[c];
            #pragma unroll
            for (int j = 0; j < 3; ++j) {
                int lt = l + (j - 1);
                if (lt >= 0 && lt < SEQLEN) {
                    float xc = xz[((size_t)b * SEQLEN + lt) * XZ_LD + 4 * D_INNER + D_MODEL + c];
                    acc += xc * lw[c * 3 + j];
                }
            }
            float gelu = 0.5f * acc * (1.f + erff(acc * 0.70710678118654752f));
            float phi  = fmaxf(0.f, gw[c] * xc_mean[b * D_MODEL + c] + gb[c]);
            val = phi * gelu;
        }
        vals[cc] = val;
        s += val; q += val * val;
    }
    rs[tid] = s; rq[tid] = q;
    __syncthreads();
    for (int off = 128; off > 0; off >>= 1) {
        if (tid < off) { rs[tid] += rs[tid + off]; rq[tid] += rq[tid + off]; }
        __syncthreads();
    }
    const float mu  = rs[0] * (1.f / CAT_LD);
    const float var = rq[0] * (1.f / CAT_LD) - mu * mu;
    const float inv = rsqrtf(var + 1e-5f);
    #pragma unroll
    for (int i = 0; i < 8; ++i) {
        int cc = tid + 256 * i;
        float o = (vals[cc] - mu) * inv * ln_g[cc] + ln_b[cc];
        size_t oo = ((size_t)b * SEQLEN + l) * CAT_LD + cc;
        bf16_t h, lo; splitf(o, h, lo);
        cat_hi[oo] = h; cat_lo[oo] = lo;
    }
}

// ---------------------------------------------------------------------------
extern "C" void kernel_launch(void* const* d_in, const int* in_sizes, int n_in,
                              void* d_out, int out_size, void* d_ws, size_t ws_size,
                              hipStream_t stream) {
    const float* hs   = (const float*)d_in[0];
    const float* inw  = (const float*)d_in[1];
    const float* cw   = (const float*)d_in[2];
    const float* cbb  = (const float*)d_in[3];
    const float* xpw  = (const float*)d_in[4];
    const float* dtw  = (const float*)d_in[5];
    const float* dtb  = (const float*)d_in[6];
    const float* Alog = (const float*)d_in[7];
    const float* Dp   = (const float*)d_in[8];
    const float* opw  = (const float*)d_in[9];
    const float* lw   = (const float*)d_in[10];
    const float* lb   = (const float*)d_in[11];
    const float* gw   = (const float*)d_in[12];
    const float* gb   = (const float*)d_in[13];
    const float* pw   = (const float*)d_in[14];
    const float* pb   = (const float*)d_in[15];
    const float* dilw = (const float*)d_in[16];
    const float* dilb = (const float*)d_in[17];
    const float* lng  = (const float*)d_in[18];
    const float* lnb  = (const float*)d_in[19];
    const float* fw   = (const float*)d_in[20];
    const float* fb   = (const float*)d_in[21];
    float* out = (float*)d_out;

    // ---- workspace bump allocator ----
    char* ws = (char*)d_ws;
    size_t off = 0;
    auto alloc = [&](size_t bytes) -> void* {
        void* p = ws + off;
        off = (off + bytes + 255) & ~(size_t)255;
        return p;
    };
    const size_t HS   = (size_t)BATCH * SEQLEN * D_MODEL;
    const size_t INW  = (size_t)5 * D_INNER * D_MODEL;
    const size_t XPW  = (size_t)(DT_RANK + 2 * D_STATE) * D_INNER;
    const size_t DTW  = (size_t)D_INNER * DT_RANK;
    const size_t OPW  = (size_t)D_MODEL * OUTF_LD;
    const size_t FW   = (size_t)(2 * D_MODEL) * CAT_LD;
    const size_t XZ   = (size_t)BATCH * SEQLEN * XZ_LD;
    const size_t XACT = (size_t)NDIR * SEQLEN * D_INNER;
    const size_t XDBL = (size_t)NDIR * SEQLEN * 64;
    const size_t OUTF = (size_t)BATCH * SEQLEN * OUTF_LD;
    const size_t CAT  = (size_t)BATCH * SEQLEN * CAT_LD;

    bf16_t *hs_h  = (bf16_t*)alloc(HS*2),   *hs_l  = (bf16_t*)alloc(HS*2);
    bf16_t *inw_h = (bf16_t*)alloc(INW*2),  *inw_l = (bf16_t*)alloc(INW*2);
    bf16_t *xpw_h = (bf16_t*)alloc(XPW*2),  *xpw_l = (bf16_t*)alloc(XPW*2);
    bf16_t *dtw_h = (bf16_t*)alloc(DTW*2),  *dtw_l = (bf16_t*)alloc(DTW*2);
    bf16_t *opw_h = (bf16_t*)alloc(OPW*2),  *opw_l = (bf16_t*)alloc(OPW*2);
    bf16_t *fw_h  = (bf16_t*)alloc(FW*2),   *fw_l  = (bf16_t*)alloc(FW*2);
    float  *xz    = (float*) alloc(XZ*4);
    float  *xact  = (float*) alloc(XACT*4);
    bf16_t *xa_h  = (bf16_t*)alloc(XACT*2), *xa_l  = (bf16_t*)alloc(XACT*2);
    float  *xdbl  = (float*) alloc(XDBL*4);
    bf16_t *xd_h  = (bf16_t*)alloc(XDBL*2), *xd_l  = (bf16_t*)alloc(XDBL*2);
    float  *draw  = (float*) alloc(XACT*4);
    bf16_t *of_h  = (bf16_t*)alloc(OUTF*2), *of_l  = (bf16_t*)alloc(OUTF*2);
    bf16_t *cat_h = (bf16_t*)alloc(CAT*2),  *cat_l = (bf16_t*)alloc(CAT*2);
    float  *xcm   = (float*) alloc(BATCH * D_MODEL * 4);
    (void)ws_size; (void)n_in; (void)in_sizes; (void)out_size;

    auto split = [&](const float* src, bf16_t* h, bf16_t* l, size_t n) {
        split_kernel<<<(unsigned)((n + 255) / 256), 256, 0, stream>>>(src, h, l, (int)n);
    };
    split(hs,  hs_h,  hs_l,  HS);
    split(inw, inw_h, inw_l, INW);
    split(xpw, xpw_h, xpw_l, XPW);
    split(dtw, dtw_h, dtw_l, DTW);
    split(opw, opw_h, opw_l, OPW);
    split(fw,  fw_h,  fw_l,  FW);

    const int NT = SEQLEN / 64;   // 36

    // GEMM1: xz(b,l,5120) = in_proj_w (5120x512) x hs(b,l,512)
    gemm_bf16x3_kernel<8><<<dim3(NT, XZ_LD/128, BATCH), 256, 0, stream>>>(
        inw_h, inw_l, D_MODEL,
        hs_h, hs_l, D_MODEL, (size_t)SEQLEN * D_MODEL,
        xz, XZ_LD, (size_t)SEQLEN * XZ_LD,
        nullptr, nullptr, 0, 0, nullptr, D_MODEL);

    // depthwise causal conv + SiLU -> x_act (8,l,1024) f32 + hi/lo
    conv_prep_kernel<<<dim3(SEQLEN, NDIR), 256, 0, stream>>>(
        xz, cw, cbb, xact, xa_h, xa_l);

    // GEMM2: x_dbl(8,l,64) = x_proj_w (64x1024) x x_act   (M=64 -> 4-wave tile)
    gemm_bf16x3_kernel<4><<<dim3(NT, 1, NDIR), 128, 0, stream>>>(
        xpw_h, xpw_l, D_INNER,
        xa_h, xa_l, D_INNER, (size_t)SEQLEN * D_INNER,
        xdbl, 64, (size_t)SEQLEN * 64,
        xd_h, xd_l, 64, (size_t)SEQLEN * 64, nullptr, D_INNER);

    // GEMM3: delta_raw(8,l,1024) = dt_proj_w (1024x32) x dt (cols 0:32 of x_dbl)
    gemm_bf16x3_kernel<8><<<dim3(NT, D_INNER/128, NDIR), 256, 0, stream>>>(
        dtw_h, dtw_l, DT_RANK,
        xd_h, xd_l, 64, (size_t)SEQLEN * 64,
        draw, D_INNER, (size_t)SEQLEN * D_INNER,
        nullptr, nullptr, 0, 0, nullptr, DT_RANK);

    // selective scan + gating -> out_f cols 0:2048 (bf16 hi/lo)
    scan_kernel<<<dim3(D_INNER/256, NDIR), 256, 0, stream>>>(
        draw, dtb, xact, xdbl, Alog, Dp, xz, of_h, of_l);

    // conv branch
    xc_mean_kernel<<<BATCH, D_MODEL, 0, stream>>>(xz, xcm);
    cb_cat_ln_kernel<<<dim3(SEQLEN, BATCH), 256, 0, stream>>>(
        xz, lw, lb, gw, gb, pw, pb, dilw, dilb, lng, lnb, xcm, cat_h, cat_l);

    // GEMM5: out_c -> out_f cols 2048:3072 (bf16 hi/lo), + fuse bias
    gemm_bf16x3_kernel<8><<<dim3(NT, (2*D_MODEL)/128, BATCH), 256, 0, stream>>>(
        fw_h, fw_l, CAT_LD,
        cat_h, cat_l, CAT_LD, (size_t)SEQLEN * CAT_LD,
        nullptr, 0, 0,
        of_h + CAT_LD, of_l + CAT_LD, OUTF_LD, (size_t)SEQLEN * OUTF_LD,
        fb, CAT_LD);

    // GEMM4: final out(b,l,512) = out_proj_w (512x3072) x out_f
    gemm_bf16x3_kernel<8><<<dim3(NT, D_MODEL/128, BATCH), 256, 0, stream>>>(
        opw_h, opw_l, OUTF_LD,
        of_h, of_l, OUTF_LD, (size_t)SEQLEN * OUTF_LD,
        out, D_MODEL, (size_t)SEQLEN * D_MODEL,
        nullptr, nullptr, 0, 0, nullptr, OUTF_LD);
}
